// DistanceAwareAttention_34806414967070
// MI455X (gfx1250) — compile-verified, matrix operand order checked by reading, not run on hardware
//
#include <hip/hip_runtime.h>

// ---------------------------------------------------------------------------
// DistanceAwareAttention (MHA + ALiBi-like distance bias + residual + LN)
// for MI455X / gfx1250: wave32, WMMA bf16 16x16x32 (f32 accum), TDM staging.
//
// Pipeline (all on `stream`, scratch in d_ws):
//   1) cvt f32->bf16: x, w_qkv, w_out
//   2) WMMA GEMM (TDM double-buffered LDS staging):
//        qkv[N,3E] = x[N,E] @ w_qkv[3E,E]^T + b_qkv                (bf16 out)
//   3) flash attention (transposed-scores trick, bias fused)        (bf16 out)
//   4) WMMA GEMM: o[N,E] = att[N,E] @ w_out[E,E]^T + b_out          (f32 out)
//   5) LayerNorm(x + o) * gamma + beta -> d_out (f32)
// ---------------------------------------------------------------------------

typedef __bf16 bf16_t;
typedef __attribute__((ext_vector_type(16))) __bf16 v16bf;
typedef __attribute__((ext_vector_type(8)))  __bf16 v8bf;
typedef __attribute__((ext_vector_type(8)))  float  v8f;
typedef __attribute__((ext_vector_type(8)))  float  f32x8;
typedef __attribute__((ext_vector_type(4)))  unsigned int u32x4;
typedef __attribute__((ext_vector_type(4)))  int i32x4;
typedef __attribute__((ext_vector_type(8)))  int i32x8;

static constexpr int Bb = 4;
static constexpr int Ss = 2048;
static constexpr int Ee = 1024;
static constexpr int Hh = 16;
static constexpr int Dd = 64;
static constexpr int Nn = Bb * Ss;   // 8192 rows total
static constexpr int E3 = 3 * Ee;    // 3072

__device__ __forceinline__ v8f wmma_bf16(v16bf a, v16bf b, v8f c) {
  // D = A(16x32) * B(32x16) + C, f32 accumulate
  return __builtin_amdgcn_wmma_f32_16x16x32_bf16(false, a, false, b, (short)0, c,
                                                 false, false);
}

// Load a 16x32 bf16 operand fragment from a row-major [16 x >=32] tile.
// ISA 7.12.2 (16-bit A 16x32): lane<16 holds row (lane&15), K={0..7,16..23};
// lane>=16 holds K={8..15,24..31}. B operand assumed symmetric (lane = column).
__device__ __forceinline__ v16bf load_frag(const bf16_t* __restrict__ base, int ld) {
  const int lane = threadIdx.x & 31;
  const bf16_t* p = base + (lane & 15) * ld + ((lane & 16) ? 8 : 0);
  v8bf lo = *(const v8bf*)(p);
  v8bf hi = *(const v8bf*)(p + 16);
  return __builtin_shufflevector(lo, hi, 0, 1, 2, 3, 4, 5, 6, 7,
                                 8, 9, 10, 11, 12, 13, 14, 15);
}

// ---------------------------------------------------------------------------
// TDM: issue a 2-D tile load (rows x 64 bf16 cols) from a row-major tensor
// with row stride Kld (elements) into LDS, with LDS row padding 64->72
// elements (pad_enable, 32 data DWORDs + 4 pad DWORDs) to match sA/sB layout.
// D# packing per CDNA5 ISA sec 8.3/8.4. Must be issued wave-uniformly.
// ---------------------------------------------------------------------------
__device__ __forceinline__ void tdm_load_tile(const bf16_t* gptr,
                                              unsigned lds_byte_off,
                                              int tile_rows, int tensor_rows,
                                              int Kld) {
  const unsigned long long ga = (unsigned long long)(size_t)gptr;
  u32x4 g0;
  g0[0] = 1u;                                   // count=1, user descriptor
  g0[1] = lds_byte_off;                         // lds_addr (bytes)
  g0[2] = (unsigned)(ga & 0xFFFFFFFFu);         // global_addr[31:0]
  g0[3] = (unsigned)((ga >> 32) & 0x1FFFFFFu)   // global_addr[56:32]
          | (2u << 30);                         // type = 2 ("image")
  const unsigned td0 = (unsigned)Kld;           // tensor_dim0 (elements)
  const unsigned td1 = (unsigned)tensor_rows;   // tensor_dim1
  i32x8 g1;
  g1[0] = (int)((1u << 16)      // data_size = 2 bytes (bf16)
                | (1u << 20)    // pad_enable
                | (4u << 22)    // pad_interval: 32 DWORDs (=64 bf16) per row
                | (3u << 25));  // pad_amount:   4 DWORDs (=8 bf16) pad
  g1[1] = (int)((td0 & 0xFFFFu) << 16);                                 // dim0 lo
  g1[2] = (int)(((td0 >> 16) & 0xFFFFu) | ((td1 & 0xFFFFu) << 16));     // dim0 hi|dim1 lo
  g1[3] = (int)(((td1 >> 16) & 0xFFFFu) | (64u << 16));                 // dim1 hi|tile_dim0=64
  g1[4] = (int)(unsigned)tile_rows;             // tile_dim1, tile_dim2=0
  g1[5] = (int)(unsigned)Kld;                   // tensor_dim0_stride[31:0]
  g1[6] = 0;                                    // stride hi | dim1_stride lo
  g1[7] = 0;
  const i32x4 z4 = {0, 0, 0, 0};
#if defined(__clang_major__) && (__clang_major__ >= 23)
  const i32x8 z8 = {0, 0, 0, 0, 0, 0, 0, 0};
  __builtin_amdgcn_tensor_load_to_lds(g0, g1, z4, z4, z8, 0);
#else
  __builtin_amdgcn_tensor_load_to_lds(g0, g1, z4, z4, 0);
#endif
}

// ---------------------------------------------------------------------------
// f32 -> bf16 conversion
// ---------------------------------------------------------------------------
__global__ void cvt_bf16_kernel(const float* __restrict__ in,
                                bf16_t* __restrict__ out, int n) {
  int i = blockIdx.x * blockDim.x + threadIdx.x;
  const int stride = gridDim.x * blockDim.x;
  for (; i < n; i += stride) out[i] = (bf16_t)in[i];
}

// ---------------------------------------------------------------------------
// Tiled WMMA GEMM:  C[M,N] = A[M,K] @ Bw[N,K]^T + bias[N]
// Block: 256 threads = 8 waves (2x4); block tile 64x128, BK=64.
// LDS tiles double-buffered; staging done by the TDM (wave 0 issues the
// descriptors, s_wait_tensorcnt synchronizes, barrier publishes to block).
// Each wave computes 32x32 of C = 2x2 WMMA tiles.
// ---------------------------------------------------------------------------
template <bool OUT_BF16>
__global__ __launch_bounds__(256) void gemm_wmma_kernel(
    const bf16_t* __restrict__ A, const bf16_t* __restrict__ Bw,
    const float* __restrict__ bias, void* __restrict__ Cout,
    int M, int N, int K) {
  constexpr int BM = 64, BN = 128, BK = 64, LDT = BK + 8;
  __shared__ bf16_t sA[2][BM][LDT];
  __shared__ bf16_t sB[2][BN][LDT];

  const int bm = blockIdx.x * BM;
  const int bn = blockIdx.y * BN;
  const int tid = threadIdx.x;
  const int lane = tid & 31;
  const int wm = (tid >> 7) & 1;  // wave row (0..1)
  const int wn = (tid >> 5) & 3;  // wave col (0..3)

  const bf16_t* Abase = A + (size_t)bm * K;
  const bf16_t* Bbase = Bw + (size_t)bn * K;

  // prologue: TDM-load first K tiles into buffer 0
  if (tid < 32) {
    tdm_load_tile(Abase, (unsigned)(size_t)&sA[0][0][0], BM, M, K);
    tdm_load_tile(Bbase, (unsigned)(size_t)&sB[0][0][0], BN, N, K);
  }

  v8f acc[2][2] = {};
  const int nk = K / BK;
  for (int ki = 0; ki < nk; ++ki) {
    const int buf = ki & 1;
    if (tid < 32) {
      __builtin_amdgcn_s_wait_tensorcnt(0);  // current buffers landed
      if (ki + 1 < nk) {                     // stream next tiles into buf^1
        tdm_load_tile(Abase + (ki + 1) * BK,
                      (unsigned)(size_t)&sA[buf ^ 1][0][0], BM, M, K);
        tdm_load_tile(Bbase + (ki + 1) * BK,
                      (unsigned)(size_t)&sB[buf ^ 1][0][0], BN, N, K);
      }
    }
    __syncthreads();  // publish current buffers to all waves
#pragma unroll
    for (int ks = 0; ks < 2; ++ks) {
      v16bf af[2], bfr[2];
#pragma unroll
      for (int i = 0; i < 2; ++i)
        af[i] = load_frag(&sA[buf][wm * 32 + i * 16][ks * 32], LDT);
#pragma unroll
      for (int j = 0; j < 2; ++j)
        bfr[j] = load_frag(&sB[buf][wn * 32 + j * 16][ks * 32], LDT);
#pragma unroll
      for (int i = 0; i < 2; ++i)
#pragma unroll
        for (int j = 0; j < 2; ++j)
          acc[i][j] = wmma_bf16(af[i], bfr[j], acc[i][j]);
    }
    __syncthreads();  // all reads done before this buffer is re-targeted
  }

  // C 16x16 f32 layout: lane = column (lane&15), VGPR r = row r + (lane&16?8:0)
  const int moff = (lane & 16) ? 8 : 0;
#pragma unroll
  for (int i = 0; i < 2; ++i) {
#pragma unroll
    for (int j = 0; j < 2; ++j) {
      const int ncol = bn + wn * 32 + j * 16 + (lane & 15);
      const float bv = bias ? bias[ncol] : 0.f;
#pragma unroll
      for (int r = 0; r < 8; ++r) {
        const int mrow = bm + wm * 32 + i * 16 + moff + r;
        const float v = acc[i][j][r] + bv;
        if constexpr (OUT_BF16)
          ((bf16_t*)Cout)[(size_t)mrow * N + ncol] = (bf16_t)v;
        else
          ((float*)Cout)[(size_t)mrow * N + ncol] = v;
      }
    }
  }
}

// ---------------------------------------------------------------------------
// Flash attention, one block (128 thr = 4 waves) per (b, h, 64 q rows).
// Wave handles a 16-row q tile. Scores computed transposed: T = K_j @ Q^T,
// so T's C-register layout IS the B-operand layout of the P@V WMMA.
// Online softmax per lane-pair (lane n and n+16 share q = n&15).
// V^T chunk staged in LDS once per block per 32-key step.
// ---------------------------------------------------------------------------
__global__ __launch_bounds__(128) void attn_kernel(
    const bf16_t* __restrict__ qkv,   // [Nn, 3E] rows: q|k|v packed
    const float* __restrict__ dist,   // [S,S]
    const float* __restrict__ lamp,   // scalar lambda
    bf16_t* __restrict__ out) {       // [Nn, E] head-interleaved
  __shared__ bf16_t vt[Dd][40];       // V^T chunk [d][key], padded to 40

  const int tid = threadIdx.x;
  const int wave = tid >> 5;
  const int lane = tid & 31;

  int ix = blockIdx.x;
  const int qb = ix & 31; ix >>= 5;   // S/64 = 32 q-blocks
  const int h  = ix & 15; ix >>= 4;   // H = 16
  const int b  = ix;                  // B = 4

  const size_t row0 = (size_t)b * Ss;
  const bf16_t* Qb = qkv + (row0 + qb * 64 + wave * 16) * E3 + h * Dd;
  const bf16_t* Kb = qkv + row0 * E3 + Ee + h * Dd;
  const bf16_t* Vb = qkv + row0 * E3 + 2 * Ee + h * Dd;

  const v16bf qf0 = load_frag(Qb, E3);        // d = 0..31
  const v16bf qf1 = load_frag(Qb + 32, E3);   // d = 32..63

  const float lamv = fabsf(lamp[0]);
  const int qrow = qb * 64 + wave * 16 + (lane & 15);
  const float* distq = dist + (size_t)qrow * Ss;
  const int koff = (lane & 16) ? 8 : 0;

  float m = -3.0e38f, l = 0.f;
  v8f acc[4] = {};  // O^T accumulators, d-tiles of 16

  for (int j = 0; j < Ss; j += 32) {
    __syncthreads();  // protect vt reads of previous iteration
    {                 // transpose-stage V[j..j+32)[0..64) -> vt[d][key]
      const int t = tid * 16;
      const int kr = t >> 6;   // key row in chunk
      const int dc = t & 63;   // d base
      const bf16_t* g = Vb + (size_t)(j + kr) * E3 + dc;
      v8bf a0 = *(const v8bf*)(g);
      v8bf a1 = *(const v8bf*)(g + 8);
#pragma unroll
      for (int e = 0; e < 8; ++e) vt[dc + e][kr] = a0[e];
#pragma unroll
      for (int e = 0; e < 8; ++e) vt[dc + 8 + e][kr] = a1[e];
    }
    __syncthreads();

    // transposed score tiles: T0 = keys j..j+15, T1 = keys j+16..j+31
    const bf16_t* K0 = Kb + (size_t)j * E3;
    const bf16_t* K1 = Kb + (size_t)(j + 16) * E3;
    v8f t0 = {}, t1 = {};
    t0 = wmma_bf16(load_frag(K0, E3), qf0, t0);
    t0 = wmma_bf16(load_frag(K0 + 32, E3), qf1, t0);
    t1 = wmma_bf16(load_frag(K1, E3), qf0, t1);
    t1 = wmma_bf16(load_frag(K1 + 32, E3), qf1, t1);

    // t0[r] = score(key = j + koff + r, q = qrow); add scale + distance bias
    const f32x8 d0 = *(const f32x8*)(distq + j + koff);
    const f32x8 d1 = *(const f32x8*)(distq + j + 16 + koff);
    float s0[8], s1[8];
    float mloc = -3.0e38f;
#pragma unroll
    for (int r = 0; r < 8; ++r) {
      s0[r] = t0[r] * 0.125f - lamv * d0[r];  // 1/sqrt(64) = 0.125
      s1[r] = t1[r] * 0.125f - lamv * d1[r];
      mloc = fmaxf(mloc, fmaxf(s0[r], s1[r]));
    }
    mloc = fmaxf(mloc, __shfl_xor(mloc, 16, 32));  // combine lane-pair halves
    const float mnew = fmaxf(m, mloc);
    const float corr = __expf(m - mnew);
    float lloc = 0.f;
    v16bf pf;  // B-operand fragment of P for the P@V WMMA (no lane swaps!)
#pragma unroll
    for (int r = 0; r < 8; ++r) {
      const float e0 = __expf(s0[r] - mnew);
      const float e1 = __expf(s1[r] - mnew);
      lloc += e0 + e1;
      pf[r] = (bf16_t)e0;      // K 0..7 (or 8..15 in upper lanes)
      pf[r + 8] = (bf16_t)e1;  // K 16..23 (or 24..31)
    }
    lloc += __shfl_xor(lloc, 16, 32);
    l = l * corr + lloc;
    m = mnew;
#pragma unroll
    for (int dt = 0; dt < 4; ++dt)
#pragma unroll
      for (int r = 0; r < 8; ++r) acc[dt][r] *= corr;

    // O^T += V^T_tile(16d x 32k) @ P(32k x 16q)
#pragma unroll
    for (int dt = 0; dt < 4; ++dt) {
      v16bf vf = load_frag(&vt[dt * 16][0], 40);
      acc[dt] = wmma_bf16(vf, pf, acc[dt]);
    }
  }

  // acc C layout: lane -> q = qrow; VGPR r -> d = dt*16 + koff + r
  const float rl = 1.f / l;
  bf16_t* orow = out + (row0 + qrow) * Ee + h * Dd;
#pragma unroll
  for (int dt = 0; dt < 4; ++dt)
#pragma unroll
    for (int r = 0; r < 8; ++r)
      orow[dt * 16 + koff + r] = (bf16_t)(acc[dt][r] * rl);
}

// ---------------------------------------------------------------------------
// y = LayerNorm(x + o) * gamma + beta ; one block (256 thr) per row of 1024
// ---------------------------------------------------------------------------
__global__ __launch_bounds__(256) void ln_kernel(
    const float* __restrict__ x, const float* __restrict__ o,
    const float* __restrict__ gamma, const float* __restrict__ beta,
    float* __restrict__ y) {
  const int row = blockIdx.x;
  const float* xr = x + (size_t)row * Ee;
  const float* orr = o + (size_t)row * Ee;
  float* yr = y + (size_t)row * Ee;

  float v[4];
  float s = 0.f, s2 = 0.f;
#pragma unroll
  for (int i = 0; i < 4; ++i) {
    const int c = threadIdx.x + i * 256;
    const float t = xr[c] + orr[c];
    v[i] = t; s += t; s2 += t * t;
  }
#pragma unroll
  for (int off = 16; off > 0; off >>= 1) {
    s += __shfl_xor(s, off, 32);
    s2 += __shfl_xor(s2, off, 32);
  }
  __shared__ float rs[8], rs2[8];
  __shared__ float mu_s, inv_s;
  const int wave = threadIdx.x >> 5, lane = threadIdx.x & 31;
  if (lane == 0) { rs[wave] = s; rs2[wave] = s2; }
  __syncthreads();
  if (threadIdx.x == 0) {
    float S = 0.f, S2 = 0.f;
#pragma unroll
    for (int i = 0; i < 8; ++i) { S += rs[i]; S2 += rs2[i]; }
    const float mu = S * (1.f / Ee);
    const float var = S2 * (1.f / Ee) - mu * mu;
    mu_s = mu;
    inv_s = rsqrtf(var + 1e-5f);
  }
  __syncthreads();
  const float mu = mu_s, inv = inv_s;
#pragma unroll
  for (int i = 0; i < 4; ++i) {
    const int c = threadIdx.x + i * 256;
    yr[c] = (v[i] - mu) * inv * gamma[c] + beta[c];
  }
}

// ---------------------------------------------------------------------------
extern "C" void kernel_launch(void* const* d_in, const int* in_sizes, int n_in,
                              void* d_out, int out_size, void* d_ws,
                              size_t ws_size, hipStream_t stream) {
  (void)in_sizes; (void)n_in; (void)out_size; (void)ws_size;
  const float* x     = (const float*)d_in[0];
  const float* dist  = (const float*)d_in[1];
  const float* lam   = (const float*)d_in[2];
  const float* wqkv  = (const float*)d_in[3];
  const float* bqkv  = (const float*)d_in[4];
  const float* wout  = (const float*)d_in[5];
  const float* bout  = (const float*)d_in[6];
  const float* gamma = (const float*)d_in[7];
  const float* beta  = (const float*)d_in[8];
  float* y = (float*)d_out;

  char* ws = (char*)d_ws;
  size_t off = 0;
  bf16_t* xb    = (bf16_t*)(ws + off); off += (size_t)Nn * Ee * 2;  // 16 MB
  bf16_t* wqkvb = (bf16_t*)(ws + off); off += (size_t)E3 * Ee * 2;  //  6 MB
  bf16_t* woutb = (bf16_t*)(ws + off); off += (size_t)Ee * Ee * 2;  //  2 MB
  bf16_t* qkvb  = (bf16_t*)(ws + off); off += (size_t)Nn * E3 * 2;  // 48 MB
  bf16_t* attb  = (bf16_t*)(ws + off); off += (size_t)Nn * Ee * 2;  // 16 MB
  float*  ofp   = (float*)(ws + off);  off += (size_t)Nn * Ee * 4;  // 32 MB

  cvt_bf16_kernel<<<2048, 256, 0, stream>>>(x, xb, Nn * Ee);
  cvt_bf16_kernel<<<2048, 256, 0, stream>>>(wqkv, wqkvb, E3 * Ee);
  cvt_bf16_kernel<<<1024, 256, 0, stream>>>(wout, woutb, Ee * Ee);

  gemm_wmma_kernel<true><<<dim3(Nn / 64, E3 / 128), 256, 0, stream>>>(
      xb, wqkvb, bqkv, qkvb, Nn, E3, Ee);

  attn_kernel<<<Bb * Hh * (Ss / 64), 128, 0, stream>>>(qkvb, dist, lam, attb);

  gemm_wmma_kernel<false><<<dim3(Nn / 64, Ee / 128), 256, 0, stream>>>(
      attb, woutb, bout, ofp, Nn, Ee, Ee);

  ln_kernel<<<Nn, 256, 0, stream>>>(x, ofp, gamma, beta, y);
}
